// MultiHeadAttention_27616639713827
// MI455X (gfx1250) — matrix operand order repaired and checked by reading, not compiled, not run on hardware
//
#include <hip/hip_runtime.h>
#include <hip/hip_bf16.h>

#define EMBED 1024
#define SEQ   2048
#define BATCH 4
#define HEADS 16
#define HDIM  64

typedef __attribute__((ext_vector_type(16))) _Float16 v16h;
typedef __attribute__((ext_vector_type(8)))  _Float16 v8h;
typedef __attribute__((ext_vector_type(8)))  float    v8f;

__device__ __forceinline__ v16h combine8(v8h lo, v8h hi) {
  v16h v;
#pragma unroll
  for (int j = 0; j < 8; ++j) { v[j] = lo[j]; v[j + 8] = hi[j]; }
  return v;
}

// A fragment (16x32 f16, MxK) from row-major matrix, row stride ld (elements).
// Lanes 0-15: row = lane, K = {k0..k0+7, k0+16..k0+23}
// Lanes 16-31: row = lane-16, K = {k0+8..k0+15, k0+24..k0+31}
__device__ __forceinline__ v16h load_frag_a(const _Float16* base, int ld, int lane, int k0) {
  const int half16 = lane & 15;
  const int koff = (lane & 16) ? 8 : 0;
  const _Float16* p = base + (size_t)half16 * ld + k0 + koff;
  return combine8(*(const v8h*)p, *(const v8h*)(p + 16));
}

// B fragment (32x16 f16, KxN) where memory holds Bmem[n][k] row-major (NT gemm):
// Lanes 0-15: col n = lane, K = {k0..k0+7, k0+16..k0+23} ; lanes 16-31: n = lane-16, K = {k0+8..k0+15, k0+24..k0+31}
__device__ __forceinline__ v16h load_frag_b(const _Float16* base, int ld, int lane, int k0) {
  const int half16 = lane & 15;
  const int koff = (lane & 16) ? 8 : 0;
  const _Float16* p = base + (size_t)half16 * ld + k0 + koff;
  return combine8(*(const v8h*)p, *(const v8h*)(p + 16));
}

__device__ __forceinline__ v8f wmma_f16(v16h a, v16h b, v8f c) {
  return __builtin_amdgcn_wmma_f32_16x16x32_f16(false, a, false, b, (short)0, c, false, false);
}

__global__ void pack_f16(const float* __restrict__ src, _Float16* __restrict__ dst, int n) {
  int i = blockIdx.x * blockDim.x + threadIdx.x;
  const int stride = gridDim.x * blockDim.x;
  for (; i < n; i += stride) dst[i] = (_Float16)src[i];
}

// C[M,N] = A[M,K] * B[N,K]^T ; 4 waves (2x2), each wave computes 64x64, block 128x128.
// MODE 0: C f16 row-major. MODE 1: f16 scattered as Vt[b][n][s] (b=m>>11, s=m&2047).
// MODE 2: C f32 row-major + bias[n].
template <int MODE>
__global__ void __launch_bounds__(128) gemm_nt(const _Float16* __restrict__ A,
                                               const _Float16* __restrict__ B,
                                               void* __restrict__ C,
                                               const float* __restrict__ bias,
                                               int M, int N, int K) {
  const int lane = threadIdx.x & 31;
  const int wave = threadIdx.x >> 5;
  const int m0 = blockIdx.x * 128 + (wave >> 1) * 64;
  const int n0 = blockIdx.y * 128 + (wave & 1) * 64;
  const int half16 = lane & 15;
  const int hi8 = (lane & 16) ? 8 : 0;

  v8f acc[4][4] = {};

  for (int k0 = 0; k0 < K; k0 += 32) {
    v16h af[4], bf[4];
#pragma unroll
    for (int im = 0; im < 4; ++im)
      af[im] = load_frag_a(A + (size_t)(m0 + im * 16) * K, K, lane, k0);
#pragma unroll
    for (int in = 0; in < 4; ++in)
      bf[in] = load_frag_b(B + (size_t)(n0 + in * 16) * K, K, lane, k0);
#pragma unroll
    for (int im = 0; im < 4; ++im)
#pragma unroll
      for (int in = 0; in < 4; ++in)
        acc[im][in] = wmma_f16(af[im], bf[in], acc[im][in]);
  }

#pragma unroll
  for (int im = 0; im < 4; ++im)
#pragma unroll
    for (int in = 0; in < 4; ++in)
#pragma unroll
      for (int r = 0; r < 8; ++r) {
        const int m = m0 + im * 16 + r + hi8;
        const int n = n0 + in * 16 + half16;
        const float v = acc[im][in][r];
        if (MODE == 0) {
          ((_Float16*)C)[(size_t)m * N + n] = (_Float16)v;
        } else if (MODE == 1) {
          const int b = m >> 11, s = m & (SEQ - 1);
          ((_Float16*)C)[((size_t)b * N + n) * SEQ + s] = (_Float16)v;
        } else {
          ((float*)C)[(size_t)m * N + n] = v + bias[n];
        }
      }
}

// Flash attention: 4 waves/block, each wave owns 32 query rows of one (b,h).
// Q,K: f16 [B*S, E] row-major. Vt: f16 [B, E, S] (so PV B-fragments are K-contiguous).
// ctx out: f16 [B*S, E].
__global__ void __launch_bounds__(128) attn_kernel(const _Float16* __restrict__ Q,
                                                   const _Float16* __restrict__ Km,
                                                   const _Float16* __restrict__ Vt,
                                                   _Float16* __restrict__ ctx) {
  __shared__ __align__(16) _Float16 lds[4][32][64];
  const int lane = threadIdx.x & 31;
  const int wave = threadIdx.x >> 5;
  const int b = blockIdx.x >> 4;
  const int h = blockIdx.x & 15;
  const int q0 = (blockIdx.y * 4 + wave) * 32;
  const int half16 = lane & 15;
  const int hi8 = (lane & 16) ? 8 : 0;

  const _Float16* Qb = Q + (size_t)(b * SEQ + q0) * EMBED + h * HDIM;
  const _Float16* Kb = Km + (size_t)b * SEQ * EMBED + h * HDIM;
  const _Float16* Vb = Vt + (size_t)b * EMBED * SEQ + (size_t)h * HDIM * SEQ;

  v16h aq[2][2];
#pragma unroll
  for (int im = 0; im < 2; ++im)
#pragma unroll
    for (int kk = 0; kk < 2; ++kk)
      aq[im][kk] = load_frag_a(Qb + (size_t)(im * 16) * EMBED, EMBED, lane, kk * 32);

  v8f o[2][4] = {};
  float rm[2][8], rs[2][8];
#pragma unroll
  for (int im = 0; im < 2; ++im)
#pragma unroll
    for (int r = 0; r < 8; ++r) { rm[im][r] = -3.0e38f; rs[im][r] = 0.0f; }

  for (int kb = 0; kb < SEQ; kb += 64) {
    // ---- scores S = Q * K^T (32 x 64 chunk) ----
    v8f sc[2][4] = {};
#pragma unroll
    for (int kk = 0; kk < 2; ++kk)
#pragma unroll
      for (int in = 0; in < 4; ++in) {
        v16h bk = load_frag_b(Kb + (size_t)(kb + in * 16) * EMBED, EMBED, lane, kk * 32);
#pragma unroll
        for (int im = 0; im < 2; ++im)
          sc[im][in] = wmma_f16(aq[im][kk], bk, sc[im][in]);
      }

    // ---- online softmax (rows live in 16-lane halves) ----
#pragma unroll
    for (int im = 0; im < 2; ++im) {
#pragma unroll
      for (int in = 0; in < 4; ++in)
#pragma unroll
        for (int r = 0; r < 8; ++r) sc[im][in][r] *= 0.125f;  // 1/sqrt(64)
#pragma unroll
      for (int r = 0; r < 8; ++r) {
        float lm = sc[im][0][r];
#pragma unroll
        for (int in = 1; in < 4; ++in) lm = fmaxf(lm, sc[im][in][r]);
#pragma unroll
        for (int msk = 1; msk < 16; msk <<= 1) lm = fmaxf(lm, __shfl_xor(lm, msk, 32));
        const float mnew = fmaxf(rm[im][r], lm);
        const float corr = __expf(rm[im][r] - mnew);
        rm[im][r] = mnew;
        float ps = 0.0f;
#pragma unroll
        for (int in = 0; in < 4; ++in) {
          const float p = __expf(sc[im][in][r] - mnew);
          sc[im][in][r] = p;
          ps += p;
        }
#pragma unroll
        for (int msk = 1; msk < 16; msk <<= 1) ps += __shfl_xor(ps, msk, 32);
        rs[im][r] = rs[im][r] * corr + ps;
#pragma unroll
        for (int in = 0; in < 4; ++in) o[im][in][r] *= corr;
      }
      // stage P (C-layout f32 -> LDS f16) for A-layout reload
#pragma unroll
      for (int in = 0; in < 4; ++in)
#pragma unroll
        for (int r = 0; r < 8; ++r)
          lds[wave][im * 16 + r + hi8][in * 16 + half16] = (_Float16)sc[im][in][r];
    }
    asm volatile("s_wait_dscnt 0x0" ::: "memory");  // DS in-order per wave; barrier for compiler too

    v16h ap[2][2];
#pragma unroll
    for (int im = 0; im < 2; ++im)
#pragma unroll
      for (int kk = 0; kk < 2; ++kk) {
        const _Float16* p = &lds[wave][im * 16 + half16][kk * 32 + ((lane & 16) ? 8 : 0)];
        ap[im][kk] = combine8(*(const v8h*)p, *(const v8h*)(p + 16));
      }

    // ---- O += P * V ----
#pragma unroll
    for (int kk = 0; kk < 2; ++kk)
#pragma unroll
      for (int id = 0; id < 4; ++id) {
        v16h bv = load_frag_b(Vb + (size_t)(id * 16) * SEQ + kb, SEQ, lane, kk * 32);
#pragma unroll
        for (int im = 0; im < 2; ++im)
          o[im][id] = wmma_f16(ap[im][kk], bv, o[im][id]);
      }
    asm volatile("s_wait_dscnt 0x0" ::: "memory");
  }

  // normalize and write context [B*S, E] f16
#pragma unroll
  for (int im = 0; im < 2; ++im)
#pragma unroll
    for (int in = 0; in < 4; ++in)
#pragma unroll
      for (int r = 0; r < 8; ++r) {
        const float v = o[im][in][r] / rs[im][r];
        ctx[(size_t)(b * SEQ + q0 + im * 16 + r + hi8) * EMBED + h * HDIM + in * 16 + half16] =
            (_Float16)v;
      }
}

extern "C" void kernel_launch(void* const* d_in, const int* in_sizes, int n_in,
                              void* d_out, int out_size, void* d_ws, size_t ws_size,
                              hipStream_t stream) {
  (void)in_sizes; (void)n_in; (void)out_size; (void)ws_size;
  const float* x  = (const float*)d_in[0];
  const float* Wq = (const float*)d_in[1];
  const float* Wk = (const float*)d_in[2];
  const float* Wv = (const float*)d_in[3];
  const float* Wo = (const float*)d_in[4];
  const float* bo = (const float*)d_in[5];

  const size_t NTOK = (size_t)BATCH * SEQ;       // 8192
  const size_t XSZ  = NTOK * EMBED;              // 8388608
  const size_t WSZ  = (size_t)EMBED * EMBED;     // 1048576

  _Float16* xh  = (_Float16*)d_ws;
  _Float16* wqh = xh  + XSZ;
  _Float16* wkh = wqh + WSZ;
  _Float16* wvh = wkh + WSZ;
  _Float16* woh = wvh + WSZ;
  _Float16* Qh  = woh + WSZ;
  _Float16* Kh  = Qh  + XSZ;
  _Float16* Vt  = Kh  + XSZ;
  _Float16* ctx = Vt  + XSZ;

  pack_f16<<<2048, 256, 0, stream>>>(x,  xh,  (int)XSZ);
  pack_f16<<<512,  256, 0, stream>>>(Wq, wqh, (int)WSZ);
  pack_f16<<<512,  256, 0, stream>>>(Wk, wkh, (int)WSZ);
  pack_f16<<<512,  256, 0, stream>>>(Wv, wvh, (int)WSZ);
  pack_f16<<<512,  256, 0, stream>>>(Wo, woh, (int)WSZ);

  dim3 gg((unsigned)(NTOK / 128), (unsigned)(EMBED / 128));  // (64, 8)
  gemm_nt<0><<<gg, 128, 0, stream>>>(xh, wqh, (void*)Qh, nullptr, (int)NTOK, EMBED, EMBED);
  gemm_nt<0><<<gg, 128, 0, stream>>>(xh, wkh, (void*)Kh, nullptr, (int)NTOK, EMBED, EMBED);
  gemm_nt<1><<<gg, 128, 0, stream>>>(xh, wvh, (void*)Vt, nullptr, (int)NTOK, EMBED, EMBED);

  attn_kernel<<<dim3(BATCH * HEADS, SEQ / 128), 128, 0, stream>>>(Qh, Kh, Vt, ctx);

  gemm_nt<2><<<gg, 128, 0, stream>>>(ctx, woh, d_out, bo, (int)NTOK, EMBED, EMBED);
}
